// ActiveSensingFramework_46969762349525
// MI455X (gfx1250) — matrix operand order repaired and checked by reading, not compiled, not run on hardware
//
#include <hip/hip_runtime.h>
#include <math.h>

#define HSZ 512
#define NSTAGES 8
#define NS 4
#define NA 64
#define NB 64
#define BSZ 2048
#define MROWS (NS * BSZ)   // 8192

#define BM 128
#define BN 128
#define BK 64
#define LDK (BK + 4)       // pad: row stride 136B keeps 8B-aligned uint2 stores

typedef __attribute__((ext_vector_type(16))) __bf16 v16bf;
typedef __attribute__((ext_vector_type(8)))  float  v8f;

union FragBF { v16bf v; unsigned u[8]; };
union FragF  { v8f   v; float    f[8]; };

__device__ __forceinline__ __bf16 f2bf(float f) {
    unsigned u = __builtin_bit_cast(unsigned, f);
    unsigned r = u + 0x7FFFu + ((u >> 16) & 1u);        // round-to-nearest-even
    unsigned short h = (unsigned short)(r >> 16);
    return __builtin_bit_cast(__bf16, h);
}

__device__ __forceinline__ float sigmoidf_(float x) { return 1.0f / (1.0f + __expf(-x)); }

// ---------------------------------------------------------------------------
// fp32 -> bf16 weight conversion (one-time; weights then live in L2)
// ---------------------------------------------------------------------------
__global__ void cvt_bf16_kernel(const float* __restrict__ in, __bf16* __restrict__ out, int n) {
    int i = blockIdx.x * blockDim.x + threadIdx.x;
    if (i < n) out[i] = f2bf(in[i]);
}

__global__ void fill_val_kernel(float* __restrict__ p, float v, int n) {
    int i = blockIdx.x * blockDim.x + threadIdx.x;
    if (i < n) p[i] = v;
}

// ---------------------------------------------------------------------------
// WMMA GEMM:  Y[M,N] = act(A[M,K] @ W[N,K]^T + bias)   (A fp32, W bf16)
// Block tile 128x128, K-tile 64, double-buffered LDS, 256 threads = 8 waves.
// Wave (wy 0..3, wx 0..1) owns a 32(M) x 64(N) sub-tile = 2x4 WMMA accumulators.
// 16 v_wmma per K-tile between barriers; next tile's global loads issued first.
// ---------------------------------------------------------------------------
__global__ __launch_bounds__(256)
void gemm_wmma_kernel(const float* __restrict__ A, const __bf16* __restrict__ W,
                      const float* __restrict__ bias, float* __restrict__ Y,
                      int M, int N, int K, int act) {
    __shared__ __bf16 sA[2][BM][LDK];
    __shared__ __bf16 sB[2][BN][LDK];

    const int tid   = threadIdx.x;
    const int lane  = tid & 31;
    const int wave  = tid >> 5;
    const int wy    = wave >> 1;      // 0..3 -> M sub-tile *32
    const int wx    = wave & 1;       // 0..1 -> N sub-tile *64
    const int lrow  = lane & 15;
    const int lhi   = lane >> 4;      // 0/1 half-wave
    const int blockM = blockIdx.y * BM;
    const int blockN = blockIdx.x * BN;

    FragF acc[2][4];
#pragma unroll
    for (int my = 0; my < 2; ++my)
#pragma unroll
        for (int nx = 0; nx < 4; ++nx)
#pragma unroll
            for (int i = 0; i < 8; ++i) acc[my][nx].f[i] = 0.0f;

    // ---- prologue: stage K-tile 0 into buffer 0 ----
#pragma unroll
    for (int it = 0; it < 8; ++it) {
        int g  = tid + it * 256;            // 2048 groups of 4 elems
        int r  = g >> 4;
        int c4 = (g & 15) << 2;
        const float4 av = *(const float4*)&A[(size_t)(blockM + r) * K + c4];
        const uint2  bv = *(const uint2*)&W[(size_t)(blockN + r) * K + c4];
        __bf16* da = &sA[0][r][c4];
        da[0] = f2bf(av.x); da[1] = f2bf(av.y); da[2] = f2bf(av.z); da[3] = f2bf(av.w);
        *(uint2*)&sB[0][r][c4] = bv;
    }
    __syncthreads();

    const int kTiles = K / BK;
    int buf = 0;
    for (int kt = 0; kt < kTiles; ++kt) {
        const bool more = (kt + 1 < kTiles);
        const int  k0n  = (kt + 1) * BK;

        // ---- issue next tile's global loads (latency hidden behind WMMAs) ----
        float4 aReg[8];
        uint2  bReg[8];
        if (more) {
#pragma unroll
            for (int it = 0; it < 8; ++it) {
                int g  = tid + it * 256;
                int r  = g >> 4;
                int c4 = (g & 15) << 2;
                aReg[it] = *(const float4*)&A[(size_t)(blockM + r) * K + k0n + c4];
                bReg[it] = *(const uint2*)&W[(size_t)(blockN + r) * K + k0n + c4];
            }
        }

        // ---- compute: 2 K-steps x (2x4) WMMAs on current buffer ----
#pragma unroll
        for (int ks = 0; ks < 2; ++ks) {
            const int kb = ks * 32;
            FragBF af[2], bfr[4];
#pragma unroll
            for (int my = 0; my < 2; ++my) {
                const int ar = wy * 32 + my * 16 + lrow;
#pragma unroll
                for (int v = 0; v < 8; ++v) {
                    // A 16x32 layout: v<4 -> K = lhi*8 + 2v ; v>=4 -> K = 16 + lhi*8 + 2(v-4)
                    const int kk = kb + ((v & 4) << 2) + lhi * 8 + 2 * (v & 3);
                    af[my].u[v] = *(const unsigned*)&sA[buf][ar][kk];
                }
            }
#pragma unroll
            for (int nx = 0; nx < 4; ++nx) {
                const int nr = wx * 64 + nx * 16 + lrow;
#pragma unroll
                for (int v = 0; v < 8; ++v) {
                    // B 32x16 layout: v<4 -> K = lhi*8 + 2v ; v>=4 -> K = 16 + lhi*8 + 2(v-4)
                    const int kk = kb + ((v & 4) << 2) + lhi * 8 + 2 * (v & 3);
                    bfr[nx].u[v] = *(const unsigned*)&sB[buf][nr][kk];
                }
            }
#pragma unroll
            for (int my = 0; my < 2; ++my)
#pragma unroll
                for (int nx = 0; nx < 4; ++nx)
                    acc[my][nx].v = __builtin_amdgcn_wmma_f32_16x16x32_bf16(
                        false, af[my].v, false, bfr[nx].v,
                        (short)0, acc[my][nx].v, false, false);
        }

        // ---- store staged registers into the other buffer, single barrier ----
        if (more) {
            const int nb = buf ^ 1;
#pragma unroll
            for (int it = 0; it < 8; ++it) {
                int g  = tid + it * 256;
                int r  = g >> 4;
                int c4 = (g & 15) << 2;
                __bf16* da = &sA[nb][r][c4];
                da[0] = f2bf(aReg[it].x); da[1] = f2bf(aReg[it].y);
                da[2] = f2bf(aReg[it].z); da[3] = f2bf(aReg[it].w);
                *(uint2*)&sB[nb][r][c4] = bReg[it];
            }
            __syncthreads();
            buf = nb;
        }
    }

    // ---- epilogue: C layout, VGPR r = row (lhi*8 + r), lane%16 = col ----
#pragma unroll
    for (int my = 0; my < 2; ++my)
#pragma unroll
        for (int nx = 0; nx < 4; ++nx) {
            const int col = blockN + wx * 64 + nx * 16 + lrow;
            const float bv = bias ? bias[col] : 0.0f;
#pragma unroll
            for (int r = 0; r < 8; ++r) {
                const int row = blockM + wy * 32 + my * 16 + lhi * 8 + r;
                float v = acc[my][nx].f[r] + bv;
                if (act == 1) v = fmaxf(v, 0.0f);
                Y[(size_t)row * N + col] = v;
            }
        }
}

// ---------------------------------------------------------------------------
// GRU elementwise fuse: h = (1-z)*n + z*h
// ---------------------------------------------------------------------------
__global__ __launch_bounds__(256)
void gru_update_kernel(const float* __restrict__ gi, const float* __restrict__ gh,
                       float* __restrict__ h) {
    int idx = blockIdx.x * blockDim.x + threadIdx.x;   // MROWS*HSZ threads
    if (idx >= MROWS * HSZ) return;
    int row = idx / HSZ, c = idx % HSZ;
    size_t base = (size_t)row * (3 * HSZ);
    float ir = gi[base + c],           hr = gh[base + c];
    float iz = gi[base + HSZ + c],     hz = gh[base + HSZ + c];
    float in = gi[base + 2 * HSZ + c], hn = gh[base + 2 * HSZ + c];
    float r = sigmoidf_(ir + hr);
    float z = sigmoidf_(iz + hz);
    float n = tanhf(in + r * hn);
    size_t ho = (size_t)row * HSZ + c;
    h[ho] = (1.0f - z) * n + z * h[ho];
}

// ---------------------------------------------------------------------------
// yb = H @ Fa + noise*sqrt(s2/2)   (per batch item; one block, 256 thr = NB*NS)
// ---------------------------------------------------------------------------
__global__ __launch_bounds__(256)
void hf_fwd_kernel(const float* __restrict__ ch, const float* __restrict__ F,
                   const float* __restrict__ noise, const float* __restrict__ sig,
                   float* __restrict__ y) {
    int b = blockIdx.x;
    __shared__ float sF[NA * NS * 2];
    for (int i = threadIdx.x; i < NA * NS * 2; i += 256)
        sF[i] = F[(size_t)b * NA * NS * 2 + i];
    __syncthreads();
    float ns = sqrtf(0.5f * sig[0]);
    int t = threadIdx.x, j = t >> 2, s = t & 3;
    const float* Hrow = &ch[((size_t)b * NB + j) * NA * 2];
    float re = 0.f, im = 0.f;
    for (int k = 0; k < NA; ++k) {
        float hr = Hrow[2 * k], hi = Hrow[2 * k + 1];
        float fr = sF[(k * NS + s) * 2], fi = sF[(k * NS + s) * 2 + 1];
        re += hr * fr - hi * fi;
        im += hr * fi + hi * fr;
    }
    size_t o = (((size_t)b * NB + j) * NS + s) * 2;
    y[o]     = re + ns * noise[o];
    y[o + 1] = im + ns * noise[o + 1];
}

// ya = H^H @ Fb + noise*sqrt(s2/2)
__global__ __launch_bounds__(256)
void hf_adj_kernel(const float* __restrict__ ch, const float* __restrict__ F,
                   const float* __restrict__ noise, const float* __restrict__ sig,
                   float* __restrict__ y) {
    int b = blockIdx.x;
    __shared__ float sF[NB * NS * 2];
    for (int i = threadIdx.x; i < NB * NS * 2; i += 256)
        sF[i] = F[(size_t)b * NB * NS * 2 + i];
    __syncthreads();
    float ns = sqrtf(0.5f * sig[0]);
    int t = threadIdx.x, k = t >> 2, s = t & 3;
    float re = 0.f, im = 0.f;
    for (int j = 0; j < NB; ++j) {
        size_t ho = (((size_t)b * NB + j) * NA + k) * 2;
        float hr = ch[ho], hi = ch[ho + 1];
        float fr = sF[(j * NS + s) * 2], fi = sF[(j * NS + s) * 2 + 1];
        re += hr * fr + hi * fi;            // conj(H)*F
        im += hr * fi - hi * fr;
    }
    size_t o = (((size_t)b * NA + k) * NS + s) * 2;
    y[o]     = re + ns * noise[o];
    y[o + 1] = im + ns * noise[o + 1];
}

// ---------------------------------------------------------------------------
// _stream_in: y (BSZ,N,NS,2) -> x rows (s*BSZ+b) x (2N)
// ---------------------------------------------------------------------------
__global__ __launch_bounds__(256)
void pack_kernel(const float* __restrict__ y, float* __restrict__ x, int N) {
    int idx = blockIdx.x * blockDim.x + threadIdx.x;   // NS*BSZ*2N
    if (idx >= NS * BSZ * 2 * N) return;
    int col = idx % (2 * N), row = idx / (2 * N);
    int s = row / BSZ, b = row % BSZ;
    int j = col >> 1, c = col & 1;
    x[idx] = y[(((size_t)b * N + j) * NS + s) * 2 + c];
}

// ---------------------------------------------------------------------------
// F = _orth(mlp_complex + y) per batch item; GS against ORIGINAL columns
// (matches reference quirk). 64 threads = one per row.
// ---------------------------------------------------------------------------
__global__ __launch_bounds__(64)
void mlp_add_orth_kernel(const float* __restrict__ mlpout, const float* __restrict__ y,
                         float* __restrict__ Fout, int N) {
    int b = blockIdx.x, j = threadIdx.x;
    __shared__ float x0[NA][NS][2];
    __shared__ float red[64 * 3];
    for (int s = 0; s < NS; ++s) {
        size_t r = ((size_t)s * BSZ + b) * (2 * N);
        size_t yo = (((size_t)b * N + j) * NS + s) * 2;
        x0[j][s][0] = mlpout[r + j]     + y[yo];
        x0[j][s][1] = mlpout[r + N + j] + y[yo + 1];
    }
    __syncthreads();
    for (int ii = 0; ii < NS; ++ii) {
        float xr = x0[j][ii][0], xi = x0[j][ii][1];
        for (int jj = 0; jj < ii; ++jj) {
            float ar = x0[j][jj][0], ai = x0[j][jj][1];
            red[j * 3 + 0] = ar * xr + ai * xi;   // Re(conj(a)*x)
            red[j * 3 + 1] = ar * xi - ai * xr;   // Im(conj(a)*x)
            red[j * 3 + 2] = ar * ar + ai * ai;   // |a|^2
            __syncthreads();
            for (int off = 32; off > 0; off >>= 1) {
                if (j < off) {
                    red[j * 3 + 0] += red[(j + off) * 3 + 0];
                    red[j * 3 + 1] += red[(j + off) * 3 + 1];
                    red[j * 3 + 2] += red[(j + off) * 3 + 2];
                }
                __syncthreads();
            }
            float cr = red[0] / red[2], ci = red[1] / red[2];
            __syncthreads();
            xr -= cr * ar - ci * ai;
            xi -= cr * ai + ci * ar;
        }
        red[j * 3 + 0] = xr * xr + xi * xi;
        red[j * 3 + 1] = 0.f; red[j * 3 + 2] = 0.f;
        __syncthreads();
        for (int off = 32; off > 0; off >>= 1) {
            if (j < off) red[j * 3 + 0] += red[(j + off) * 3 + 0];
            __syncthreads();
        }
        float inv = rsqrtf(red[0]);
        __syncthreads();
        size_t o = (((size_t)b * N + j) * NS + ii) * 2;
        Fout[o]     = xr * inv;
        Fout[o + 1] = xi * inv;
    }
}

// ---------------------------------------------------------------------------
// Fa init: Fa[b] = Wa / ||col||  broadcast over batch
// ---------------------------------------------------------------------------
__global__ __launch_bounds__(64)
void init_fa_kernel(const float* __restrict__ W, float* __restrict__ Fa) {
    int b = blockIdx.x, j = threadIdx.x;
    __shared__ float red[64];
    for (int s = 0; s < NS; ++s) {
        float wr = W[(j * NS + s) * 2], wi = W[(j * NS + s) * 2 + 1];
        red[j] = wr * wr + wi * wi;
        __syncthreads();
        for (int off = 32; off > 0; off >>= 1) {
            if (j < off) red[j] += red[j + off];
            __syncthreads();
        }
        float inv = rsqrtf(red[0]);
        __syncthreads();
        size_t o = (((size_t)b * NA + j) * NS + s) * 2;
        Fa[o] = wr * inv; Fa[o + 1] = wi * inv;
    }
}

// ---------------------------------------------------------------------------
// loss -= logdet(I + G G^H/(NS*s2))/BSZ,  G = Fb^H H Fa  (4x4 Hermitian PD)
// ---------------------------------------------------------------------------
__global__ __launch_bounds__(256)
void logdet_kernel(const float* __restrict__ ch, const float* __restrict__ Fa,
                   const float* __restrict__ Fb, const float* __restrict__ sig,
                   float* __restrict__ loss) {
    int b = blockIdx.x;
    __shared__ float sFa[NA * NS * 2], sFb[NB * NS * 2];
    __shared__ float sT[NS][NA][2];
    __shared__ float sG[NS][NS][2];
    for (int i = threadIdx.x; i < NA * NS * 2; i += 256) {
        sFa[i] = Fa[(size_t)b * NA * NS * 2 + i];
        sFb[i] = Fb[(size_t)b * NB * NS * 2 + i];
    }
    __syncthreads();
    int t = threadIdx.x, p = t >> 6, k = t & 63;
    {   // t[p][k] = sum_j conj(Fb[j][p]) * H[j][k]
        float re = 0.f, im = 0.f;
        for (int j = 0; j < NB; ++j) {
            size_t ho = (((size_t)b * NB + j) * NA + k) * 2;
            float hr = ch[ho], hi = ch[ho + 1];
            float fr = sFb[(j * NS + p) * 2], fi = sFb[(j * NS + p) * 2 + 1];
            re += fr * hr + fi * hi;
            im += fr * hi - fi * hr;
        }
        sT[p][k][0] = re; sT[p][k][1] = im;
    }
    __syncthreads();
    if (t < NS * NS) {   // G = T @ Fa
        int pp = t >> 2, q = t & 3;
        float gr = 0.f, gi = 0.f;
        for (int kk = 0; kk < NA; ++kk) {
            float tr = sT[pp][kk][0], ti = sT[pp][kk][1];
            float fr = sFa[(kk * NS + q) * 2], fi = sFa[(kk * NS + q) * 2 + 1];
            gr += tr * fr - ti * fi;
            gi += tr * fi + ti * fr;
        }
        sG[pp][q][0] = gr; sG[pp][q][1] = gi;
    }
    __syncthreads();
    if (t == 0) {
        float inv = 1.0f / (NS * sig[0]);
        float Mr[NS][NS], Mi[NS][NS];
        for (int pp = 0; pp < NS; ++pp)
            for (int q = 0; q < NS; ++q) {
                float re = 0.f, im = 0.f;
                for (int r = 0; r < NS; ++r) {
                    float ar = sG[pp][r][0], ai = sG[pp][r][1];
                    float br = sG[q][r][0],  bi = sG[q][r][1];
                    re += ar * br + ai * bi;         // G[p][r]*conj(G[q][r])
                    im += ai * br - ar * bi;
                }
                Mr[pp][q] = (pp == q ? 1.0f : 0.0f) + inv * re;
                Mi[pp][q] = inv * im;
            }
        // complex Cholesky, logdet = 2*sum log L_ii
        float Lr[NS][NS], Li[NS][NS], ld = 0.f;
        for (int i = 0; i < NS; ++i) {
            float dr = Mr[i][i];
            for (int kk = 0; kk < i; ++kk) dr -= Lr[i][kk] * Lr[i][kk] + Li[i][kk] * Li[i][kk];
            float lii = sqrtf(fmaxf(dr, 1e-30f));
            Lr[i][i] = lii; Li[i][i] = 0.f;
            ld += 2.0f * __logf(lii);
            for (int pp = i + 1; pp < NS; ++pp) {
                float sr = Mr[pp][i], si = Mi[pp][i];
                for (int kk = 0; kk < i; ++kk) {
                    sr -= Lr[pp][kk] * Lr[i][kk] + Li[pp][kk] * Li[i][kk];
                    si -= Li[pp][kk] * Lr[i][kk] - Lr[pp][kk] * Li[i][kk];
                }
                Lr[pp][i] = sr / lii; Li[pp][i] = si / lii;
            }
        }
        atomicAdd(loss, -ld * (1.0f / BSZ));
    }
}

// ---------------------------------------------------------------------------
// host orchestration
// ---------------------------------------------------------------------------
extern "C" void kernel_launch(void* const* d_in, const int* in_sizes, int n_in,
                              void* d_out, int out_size, void* d_ws, size_t ws_size,
                              hipStream_t stream) {
    (void)in_sizes; (void)n_in; (void)out_size; (void)ws_size;
    const float* channel = (const float*)d_in[0];
    const float* sigma2  = (const float*)d_in[1];
    const float* noise_b = (const float*)d_in[2];
    const float* noise_a = (const float*)d_in[3];
    const float* init_Wa = (const float*)d_in[4];
    const float* Wih_a = (const float*)d_in[5];
    const float* Whh_a = (const float*)d_in[6];
    const float* bih_a = (const float*)d_in[7];
    const float* bhh_a = (const float*)d_in[8];
    const float* Wih_b = (const float*)d_in[9];
    const float* Whh_b = (const float*)d_in[10];
    const float* bih_b = (const float*)d_in[11];
    const float* bhh_b = (const float*)d_in[12];
    const float* Wa1  = (const float*)d_in[13];  const float* ba1  = (const float*)d_in[14];
    const float* Wa2  = (const float*)d_in[15];  const float* ba2  = (const float*)d_in[16];
    const float* Wb1  = (const float*)d_in[17];  const float* bb1  = (const float*)d_in[18];
    const float* Wb2  = (const float*)d_in[19];  const float* bb2  = (const float*)d_in[20];
    const float* Wa1d = (const float*)d_in[21];  const float* ba1d = (const float*)d_in[22];
    const float* Wa2d = (const float*)d_in[23];  const float* ba2d = (const float*)d_in[24];
    const float* Wb1d = (const float*)d_in[25];  const float* bb1d = (const float*)d_in[26];
    const float* Wb2d = (const float*)d_in[27];  const float* bb2d = (const float*)d_in[28];

    char* ws = (char*)d_ws;
    size_t off = 0;
    auto carve = [&](size_t bytes) -> void* {
        void* p = ws + off;
        off = (off + bytes + 255) & ~(size_t)255;
        return p;
    };
    // bf16 weights
    __bf16* bWihA = (__bf16*)carve((size_t)3 * HSZ * 2 * NA * 2);
    __bf16* bWhhA = (__bf16*)carve((size_t)3 * HSZ * HSZ * 2);
    __bf16* bWihB = (__bf16*)carve((size_t)3 * HSZ * 2 * NB * 2);
    __bf16* bWhhB = (__bf16*)carve((size_t)3 * HSZ * HSZ * 2);
    __bf16* bWa1  = (__bf16*)carve((size_t)1024 * HSZ * 2);
    __bf16* bWa2  = (__bf16*)carve((size_t)2 * NA * 1024 * 2);
    __bf16* bWb1  = (__bf16*)carve((size_t)1024 * HSZ * 2);
    __bf16* bWb2  = (__bf16*)carve((size_t)2 * NB * 1024 * 2);
    __bf16* bWa1d = (__bf16*)carve((size_t)1024 * HSZ * 2);
    __bf16* bWa2d = (__bf16*)carve((size_t)2 * NA * 1024 * 2);
    __bf16* bWb1d = (__bf16*)carve((size_t)1024 * HSZ * 2);
    __bf16* bWb2d = (__bf16*)carve((size_t)2 * NB * 1024 * 2);
    // fp32 activations
    float* x      = (float*)carve((size_t)MROWS * 2 * NA * 4);
    float* gi     = (float*)carve((size_t)MROWS * 3 * HSZ * 4);
    float* gh     = (float*)carve((size_t)MROWS * 3 * HSZ * 4);
    float* hid    = (float*)carve((size_t)MROWS * 1024 * 4);
    float* mlpout = (float*)carve((size_t)MROWS * 2 * NA * 4);
    float* hA     = (float*)carve((size_t)MROWS * HSZ * 4);
    float* hB     = (float*)carve((size_t)MROWS * HSZ * 4);
    float* Fa     = (float*)carve((size_t)BSZ * NA * NS * 2 * 4);
    float* Fb     = (float*)carve((size_t)BSZ * NB * NS * 2 * 4);
    float* ya     = (float*)carve((size_t)BSZ * NA * NS * 2 * 4);
    float* yb     = (float*)carve((size_t)BSZ * NB * NS * 2 * 4);

    float* out   = (float*)d_out;
    float* outFa = out;                                  // (BSZ,NA,NS) complex
    float* outFb = out + (size_t)BSZ * NA * NS * 2;      // (BSZ,NB,NS) complex
    float* lossp = out + (size_t)2 * BSZ * NA * NS * 2;  // scalar

    auto cvt = [&](const float* src, __bf16* dst, int n) {
        cvt_bf16_kernel<<<(n + 255) / 256, 256, 0, stream>>>(src, dst, n);
    };
    cvt(Wih_a, bWihA, 3 * HSZ * 2 * NA);  cvt(Whh_a, bWhhA, 3 * HSZ * HSZ);
    cvt(Wih_b, bWihB, 3 * HSZ * 2 * NB);  cvt(Whh_b, bWhhB, 3 * HSZ * HSZ);
    cvt(Wa1, bWa1, 1024 * HSZ);  cvt(Wa2, bWa2, 2 * NA * 1024);
    cvt(Wb1, bWb1, 1024 * HSZ);  cvt(Wb2, bWb2, 2 * NB * 1024);
    cvt(Wa1d, bWa1d, 1024 * HSZ); cvt(Wa2d, bWa2d, 2 * NA * 1024);
    cvt(Wb1d, bWb1d, 1024 * HSZ); cvt(Wb2d, bWb2d, 2 * NB * 1024);

    fill_val_kernel<<<(MROWS * HSZ + 255) / 256, 256, 0, stream>>>(hA, 1.0f, MROWS * HSZ);
    fill_val_kernel<<<(MROWS * HSZ + 255) / 256, 256, 0, stream>>>(hB, 1.0f, MROWS * HSZ);
    fill_val_kernel<<<1, 1, 0, stream>>>(lossp, 0.0f, 1);
    init_fa_kernel<<<BSZ, 64, 0, stream>>>(init_Wa, Fa);

    auto gemm = [&](const float* A, const __bf16* W, const float* bias, float* Y,
                    int N, int K, int act) {
        dim3 grid(N / BN, MROWS / BM);
        gemm_wmma_kernel<<<grid, 256, 0, stream>>>(A, W, bias, Y, MROWS, N, K, act);
    };
    const int packBlocks = (NS * BSZ * 2 * NA + 255) / 256;
    const int gruBlocks  = (MROWS * HSZ + 255) / 256;

    for (int st = 0; st < NSTAGES; ++st) {
        const float* nb = noise_b + (size_t)st * BSZ * NB * NS * 2;
        const float* na = noise_a + (size_t)st * BSZ * NA * NS * 2;

        // --- B side: receive yb, GRU, MLP, orth ---
        hf_fwd_kernel<<<BSZ, 256, 0, stream>>>(channel, Fa, nb, sigma2, yb);
        pack_kernel<<<packBlocks, 256, 0, stream>>>(yb, x, NB);
        gemm(x,  bWihB, bih_b, gi, 3 * HSZ, 2 * NB, 0);
        gemm(hB, bWhhB, bhh_b, gh, 3 * HSZ, HSZ,    0);
        gru_update_kernel<<<gruBlocks, 256, 0, stream>>>(gi, gh, hB);
        gemm(hB,  bWb1, bb1, hid,    1024,   HSZ,  1);
        gemm(hid, bWb2, bb2, mlpout, 2 * NB, 1024, 0);
        mlp_add_orth_kernel<<<BSZ, 64, 0, stream>>>(mlpout, yb, Fb, NB);

        // --- A side: receive ya, GRU, MLP, orth ---
        hf_adj_kernel<<<BSZ, 256, 0, stream>>>(channel, Fb, na, sigma2, ya);
        pack_kernel<<<packBlocks, 256, 0, stream>>>(ya, x, NA);
        gemm(x,  bWihA, bih_a, gi, 3 * HSZ, 2 * NA, 0);
        gemm(hA, bWhhA, bhh_a, gh, 3 * HSZ, HSZ,    0);
        gru_update_kernel<<<gruBlocks, 256, 0, stream>>>(gi, gh, hA);
        gemm(hA,  bWa1, ba1, hid,    1024,   HSZ,  1);
        gemm(hid, bWa2, ba2, mlpout, 2 * NA, 1024, 0);
        mlp_add_orth_kernel<<<BSZ, 64, 0, stream>>>(mlpout, ya, Fa, NA);

        // --- detached heads -> Fb_d, Fa_d (written to d_out; last stage wins) ---
        gemm(hB,  bWb1d, bb1d, hid,    1024,   HSZ,  1);
        gemm(hid, bWb2d, bb2d, mlpout, 2 * NB, 1024, 0);
        mlp_add_orth_kernel<<<BSZ, 64, 0, stream>>>(mlpout, yb, outFb, NB);
        gemm(hA,  bWa1d, ba1d, hid,    1024,   HSZ,  1);
        gemm(hid, bWa2d, ba2d, mlpout, 2 * NA, 1024, 0);
        mlp_add_orth_kernel<<<BSZ, 64, 0, stream>>>(mlpout, ya, outFa, NA);

        // --- loss accumulation ---
        logdet_kernel<<<BSZ, 256, 0, stream>>>(channel, outFa, outFb, sigma2, lossp);
    }
}
